// MMGCN_47665547051030
// MI455X (gfx1250) — matrix operand order repaired and checked by reading, not compiled, not run on hardware
//
#include <hip/hip_runtime.h>
#include <hip/hip_bf16.h>

#define N_USERS 60000
#define N_ITEMS 30000
#define N_NODES 90000
#define DCOL    192
#define N_EDGES 500000
#define N_DIR   (2 * N_EDGES)

typedef __attribute__((ext_vector_type(16))) _Float16 v16h;
typedef __attribute__((ext_vector_type(8)))  float    v8f;

struct Edge { int s; int d; float v; };

// ---------------------------------------------------------------- utilities
__global__ void zero_kernel(float* __restrict__ p, long long n) {
    long long i = (long long)blockIdx.x * blockDim.x + threadIdx.x;
    if (i < n) p[i] = 0.0f;
}

__global__ void add_kernel(float* __restrict__ a, const float* __restrict__ b, long long n) {
    long long i = (long long)blockIdx.x * blockDim.x + threadIdx.x;
    if (i < n) a[i] += b[i];
}

// ---------------------------------------------------------------- graph prep
__global__ void degree_kernel(const long long* __restrict__ ei, float* __restrict__ deg) {
    int e = blockIdx.x * blockDim.x + threadIdx.x;
    if (e >= N_EDGES) return;
    atomicAdd(&deg[(int)ei[e]], 1.0f);
    atomicAdd(&deg[(int)ei[N_EDGES + e]], 1.0f);
}

__global__ void edge_prep_kernel(const long long* __restrict__ ei,
                                 const float* __restrict__ deg,
                                 Edge* __restrict__ edges) {
    int e = blockIdx.x * blockDim.x + threadIdx.x;
    if (e >= N_DIR) return;
    int s, d;
    if (e < N_EDGES) { s = (int)ei[e];                       d = (int)ei[N_EDGES + e]; }
    else             { s = (int)ei[N_EDGES + (e - N_EDGES)]; d = (int)ei[e - N_EDGES]; }
    float v = rsqrtf(fmaxf(deg[s], 1.0f)) * rsqrtf(fmaxf(deg[d], 1.0f));
    Edge ed; ed.s = s; ed.d = d; ed.v = v;
    edges[e] = ed;
}

// ---------------------------------------------------------------- x init (users)
__global__ void user_fill_kernel(const float* __restrict__ ue,
                                 float* __restrict__ x, float* __restrict__ acc) {
    int idx = blockIdx.x * blockDim.x + threadIdx.x;     // over 60000*192
    if (idx >= N_USERS * DCOL) return;
    int u = idx / DCOL, c = idx - u * DCOL;
    int b = c >> 6, d = c & 63;
    float v = ue[((long long)b * N_USERS + u) * 64 + d]; // user_emb[b][u][d]
    long long o = (long long)u * DCOL + c;
    x[o] = v; acc[o] = v;
}

// ---------------------------------------------------------------- item projection (WMMA)
// grid: (30000/16), block 384 (12 waves). wave w owns N-tile w in [0,12):
// b = w>>2, local d-range = (w&3)*16. All 12 waves share the LDS A tile, so the
// 169MB feature matrix is streamed from HBM exactly once (memory-roofline bound).
// A staging: 128 threads x float4 = 512 floats -> global_load_b128 + ds_store_b128.
__global__ __launch_bounds__(384)
void proj_kernel(const float* __restrict__ xt, const float* __restrict__ xi,
                 const float* __restrict__ xs,
                 const float* __restrict__ Wt, const float* __restrict__ Wi,
                 const float* __restrict__ Ws,
                 const float* __restrict__ bt, const float* __restrict__ bi,
                 const float* __restrict__ bs,
                 float* __restrict__ x, float* __restrict__ acc) {
    __shared__ float As[16 * 32];
    const int m0   = blockIdx.x * 16;
    const int tid  = threadIdx.x;
    const int wave = tid >> 5;                   // 0..11 == N-tile id
    const int lane = tid & 31;
    const int half = lane >> 4, l16 = lane & 15;
    const int bsel = wave >> 2;
    const int n0   = (wave & 3) * 16;            // col offset within D=64
    const int sr   = tid >> 3;                   // staging row   (tid<128: 0..15)
    const int sc   = (tid & 7) * 4;              // staging col   (0,4,...,28)
    v8f c = {};
    for (int seg = 0; seg < 3; ++seg) {
        const float* A; const float* W; int Km;
        if (seg == 0)      { A = xt; W = Wt + (long long)bsel * 768 * 64; Km = 768; }
        else if (seg == 1) { A = xi; W = Wi + (long long)bsel * 512 * 64; Km = 512; }
        else               { A = xs; W = Ws + (long long)bsel * 128 * 64; Km = 128; }
        for (int k0 = 0; k0 < Km; k0 += 32) {
            __syncthreads();
            if (tid < 128) {
                const float* src = &A[(long long)(m0 + sr) * Km + k0 + sc];
                *(float4*)&As[sr * 32 + sc] = *(const float4*)src;
                if (k0 + 32 < Km) __builtin_prefetch(src + 32, 0, 3); // global_prefetch_b8
            }
            __syncthreads();
            v16h af, bf;
            const float* arow = &As[l16 * 32 + half * 8];
            #pragma unroll
            for (int j = 0; j < 8; ++j) {
                af[j]     = (_Float16)arow[j];
                af[8 + j] = (_Float16)arow[16 + j];
            }
            const float* brow = W + (long long)k0 * 64 + n0 + l16;
            #pragma unroll
            for (int j = 0; j < 8; ++j) {
                bf[j]     = (_Float16)brow[(half * 8 + j) * 64];
                bf[8 + j] = (_Float16)brow[(16 + half * 8 + j) * 64];
            }
            c = __builtin_amdgcn_wmma_f32_16x16x32_f16(false, af, false, bf,
                                                       (short)0, c, false, false);
        }
    }
    const int dcol = n0 + l16;                       // 0..63
    const float bias = bt[bsel * 64 + dcol] + bi[bsel * 64 + dcol] + bs[bsel * 64 + dcol];
    const int col = bsel * 64 + dcol;                // 0..191
    #pragma unroll
    for (int r = 0; r < 8; ++r) {
        int m = m0 + r + half * 8;
        long long o = (long long)(N_USERS + m) * DCOL + col;
        float v = c[r] + bias;
        x[o] = v; acc[o] = v;
    }
}

// ---------------------------------------------------------------- SpMM (atomic scatter)
// block 192 threads, each block handles 8 directed edges; x_next[s] += v * x_cur[d]
// x_next (69MB) is L2-resident (192MB L2) so the f32 atomics resolve in L2.
__global__ __launch_bounds__(192)
void spmm_kernel(const Edge* __restrict__ edges,
                 const float* __restrict__ xin, float* __restrict__ xout) {
    const int tx = threadIdx.x;                  // feature column 0..191
    const int ebase = blockIdx.x * 8;
    #pragma unroll
    for (int i = 0; i < 8; ++i) {
        int e = ebase + i;
        if (e >= N_DIR) return;
        Edge ed = edges[e];
        float v = ed.v * xin[(long long)ed.d * DCOL + tx];
        atomicAdd(&xout[(long long)ed.s * DCOL + tx], v);
    }
}

// ---------------------------------------------------------------- attention GEMM1 (WMMA)
// H = relu((acc/3) @ W1 + b1), W1 per node type. grid (90000/16), block 256
// (8 waves; wave w owns N-tile w of 128 cols) -> acc streamed from HBM once.
__global__ __launch_bounds__(256)
void attn1_kernel(const float* __restrict__ acc,
                  const float* __restrict__ Wu1, const float* __restrict__ bu1,
                  const float* __restrict__ Wi1, const float* __restrict__ bi1,
                  float* __restrict__ H) {
    __shared__ float As[16 * 32];
    const int m0 = blockIdx.x * 16;
    const bool isUser = (m0 < N_USERS);          // 60000 % 16 == 0 -> tiles homogeneous
    const float* W1 = isUser ? Wu1 : Wi1;
    const float* b1 = isUser ? bu1 : bi1;
    const int tid  = threadIdx.x;
    const int wave = tid >> 5;                   // 0..7
    const int lane = tid & 31;
    const int half = lane >> 4, l16 = lane & 15;
    const int n0 = wave * 16;                    // 0..112
    const int sr = tid >> 3;                     // staging row (tid<128)
    const int sc = (tid & 7) * 4;                // staging col
    const float scale = 1.0f / 3.0f;
    v8f c = {};
    for (int k0 = 0; k0 < DCOL; k0 += 32) {
        __syncthreads();
        if (tid < 128) {
            float4 v4 = *(const float4*)&acc[(long long)(m0 + sr) * DCOL + k0 + sc];
            v4.x *= scale; v4.y *= scale; v4.z *= scale; v4.w *= scale;
            *(float4*)&As[sr * 32 + sc] = v4;
        }
        __syncthreads();
        v16h af, bf;
        const float* arow = &As[l16 * 32 + half * 8];
        #pragma unroll
        for (int j = 0; j < 8; ++j) {
            af[j]     = (_Float16)arow[j];
            af[8 + j] = (_Float16)arow[16 + j];
        }
        const float* brow = W1 + (long long)k0 * 128 + n0 + l16;
        #pragma unroll
        for (int j = 0; j < 8; ++j) {
            bf[j]     = (_Float16)brow[(half * 8 + j) * 128];
            bf[8 + j] = (_Float16)brow[(16 + half * 8 + j) * 128];
        }
        c = __builtin_amdgcn_wmma_f32_16x16x32_f16(false, af, false, bf,
                                                   (short)0, c, false, false);
    }
    const int ncol = n0 + l16;
    const float bias = b1[ncol];
    #pragma unroll
    for (int r = 0; r < 8; ++r) {
        int m = m0 + r + half * 8;
        H[(long long)m * 128 + ncol] = fmaxf(c[r] + bias, 0.0f);
    }
}

// ---------------------------------------------------------------- logits + softmax + pool
__global__ void attn2_out_kernel(const float* __restrict__ H, const float* __restrict__ acc,
                                 const float* __restrict__ Wu2, const float* __restrict__ bu2,
                                 const float* __restrict__ Wi2, const float* __restrict__ bi2,
                                 float* __restrict__ out) {
    int n = blockIdx.x * blockDim.x + threadIdx.x;
    if (n >= N_NODES) return;
    const bool isUser = (n < N_USERS);
    const float* W2 = isUser ? Wu2 : Wi2;
    const float* b2 = isUser ? bu2 : bi2;
    float z0 = b2[0], z1 = b2[1], z2 = b2[2];
    const float* h = H + (long long)n * 128;
    #pragma unroll 4
    for (int k = 0; k < 128; ++k) {
        float hv = h[k];
        z0 += hv * W2[k * 3 + 0];
        z1 += hv * W2[k * 3 + 1];
        z2 += hv * W2[k * 3 + 2];
    }
    float mx = fmaxf(z0, fmaxf(z1, z2));
    float e0 = __expf(z0 - mx), e1 = __expf(z1 - mx), e2 = __expf(z2 - mx);
    float inv = 1.0f / (e0 + e1 + e2);
    const float third = 1.0f / 3.0f;               // out = acc/3 weighted by softmax
    float a0 = e0 * inv * third, a1 = e1 * inv * third, a2 = e2 * inv * third;
    const float* row = acc + (long long)n * DCOL;
    float* o = out + (long long)n * 64;
    #pragma unroll 4
    for (int d = 0; d < 64; ++d)
        o[d] = a0 * row[d] + a1 * row[64 + d] + a2 * row[128 + d];
}

// ---------------------------------------------------------------- launcher
extern "C" void kernel_launch(void* const* d_in, const int* in_sizes, int n_in,
                              void* d_out, int out_size, void* d_ws, size_t ws_size,
                              hipStream_t stream) {
    (void)in_sizes; (void)n_in; (void)out_size; (void)ws_size;
    const float* x_txt  = (const float*)d_in[0];
    const float* x_img  = (const float*)d_in[1];
    const float* x_st   = (const float*)d_in[2];
    const float* u_emb  = (const float*)d_in[3];
    const float* W_txt  = (const float*)d_in[4];
    const float* b_txt  = (const float*)d_in[5];
    const float* W_img  = (const float*)d_in[6];
    const float* b_img  = (const float*)d_in[7];
    const float* W_stm  = (const float*)d_in[8];
    const float* b_stm  = (const float*)d_in[9];
    const float* Wu1    = (const float*)d_in[10];
    const float* bu1    = (const float*)d_in[11];
    const float* Wu2    = (const float*)d_in[12];
    const float* bu2    = (const float*)d_in[13];
    const float* Wi1    = (const float*)d_in[14];
    const float* bi1    = (const float*)d_in[15];
    const float* Wi2    = (const float*)d_in[16];
    const float* bi2    = (const float*)d_in[17];
    const long long* ei = (const long long*)d_in[18];
    float* out = (float*)d_out;

    const long long NX = (long long)N_NODES * DCOL;  // 17,280,000 floats
    float* ws  = (float*)d_ws;
    float* deg = ws;                                  // 90112 floats (padded)
    float* acc = ws + 90112;
    float* xa  = acc + NX;
    float* xb  = xa + NX;
    Edge*  edg = (Edge*)(xb + NX);                    // 1M directed edges * 12B

    // 1) degree + normalized edge list
    zero_kernel<<<(90112 + 255) / 256, 256, 0, stream>>>(deg, 90112);
    degree_kernel<<<(N_EDGES + 255) / 256, 256, 0, stream>>>(ei, deg);
    edge_prep_kernel<<<(N_DIR + 255) / 256, 256, 0, stream>>>(ei, deg, edg);

    // 2) build x0 (users + projected items); acc = x0
    user_fill_kernel<<<(N_USERS * DCOL + 255) / 256, 256, 0, stream>>>(u_emb, xa, acc);
    proj_kernel<<<N_ITEMS / 16, 384, 0, stream>>>(x_txt, x_img, x_st, W_txt, W_img, W_stm,
                                                  b_txt, b_img, b_stm, xa, acc);

    // 3) two rounds of propagation: acc += A x0 ; acc += A^2 x0
    int zgrid = (int)((NX + 255) / 256);
    int sgrid = (N_DIR + 7) / 8;
    zero_kernel<<<zgrid, 256, 0, stream>>>(xb, NX);
    spmm_kernel<<<sgrid, 192, 0, stream>>>(edg, xa, xb);
    add_kernel<<<zgrid, 256, 0, stream>>>(acc, xb, NX);

    zero_kernel<<<zgrid, 256, 0, stream>>>(xa, NX);
    spmm_kernel<<<sgrid, 192, 0, stream>>>(edg, xb, xa);
    add_kernel<<<zgrid, 256, 0, stream>>>(acc, xa, NX);

    // 4) attention MLP hidden layer (H reuses xb region: 90000*128 < 90000*192)
    float* H = xb;
    attn1_kernel<<<N_NODES / 16, 256, 0, stream>>>(acc, Wu1, bu1, Wi1, bi1, H);

    // 5) logits + softmax + weighted pooling -> out (90000 x 64)
    attn2_out_kernel<<<(N_NODES + 255) / 256, 256, 0, stream>>>(H, acc, Wu2, bu2, Wi2, bi2, out);
}
